// STN_6975026889033
// MI455X (gfx1250) — matrix operand order repaired and checked by reading, not compiled, not run on hardware
//
#include <hip/hip_runtime.h>
#include <hip/hip_bf16.h>
#include <math.h>

// ---------------------------------------------------------------------------
// STN forward for MI455X (gfx1250, wave32, WMMA).
//   Kernel 1: localization net (conv7x7 -> pool -> relu -> conv5x5 -> pool ->
//             relu -> fc90x32 -> relu -> fc32x6) as batch-as-M implicit GEMMs
//             on v_wmma_f32_16x16x32_bf16.  One wave per 16-image tile.
//   Kernel 2: affine grid + bilinear sampling (reflect), memory bound.
// ---------------------------------------------------------------------------

typedef __attribute__((ext_vector_type(16))) __bf16 v16bf;
typedef __attribute__((ext_vector_type(8)))  __bf16 v8bf;
typedef __attribute__((ext_vector_type(4)))  __bf16 v4bf;
typedef __attribute__((ext_vector_type(8)))  float  v8f;

__device__ __forceinline__ __bf16 f2bf(float f) { return (__bf16)f; }

// Wave-local LDS ordering: all lanes of the (single) wave execute the same
// instruction stream, so waiting DScnt==0 makes prior ds stores/loads of the
// whole wave visible/complete before the next LDS op.
__device__ __forceinline__ void lds_sync() {
  asm volatile("s_wait_dscnt 0" ::: "memory");
}

// A fragment (16x32, this code's contiguous k order, same as load_b):
//   lane L: m = L&15 ; kbase = (L<16)?0:16
//   elements e = col[m][koff + kbase + e]
// -> two 16-byte ds loads.
__device__ __forceinline__ v16bf load_a(const __bf16* row, int koff, int kbase) {
  union { v16bf v; v8bf h[2]; } u;
  u.h[0] = *reinterpret_cast<const v8bf*>(row + koff + kbase);
  u.h[1] = *reinterpret_cast<const v8bf*>(row + koff + kbase + 8);
  return u.v;
}

// B fragment (32x16, rows striped across lanes within a VGPR):
//   lane L: n = L&15 ; kb = (L<16)?0:16 ; elements e = Wt[n][koff + kb + e]
// Wt stored transposed+padded as [N][Kpad] bf16 -> 16 contiguous elements.
__device__ __forceinline__ v16bf load_b(const __bf16* row, int koff, int kb) {
  union { v16bf v; v8bf h[2]; } u;
  u.h[0] = *reinterpret_cast<const v8bf*>(row + koff + kb);
  u.h[1] = *reinterpret_cast<const v8bf*>(row + koff + kb + 8);
  return u.v;
}

__device__ __forceinline__ v8f wmma_bf16(v16bf a, v16bf b, v8f c) {
  return __builtin_amdgcn_wmma_f32_16x16x32_bf16(false, a, false, b,
                                                 (short)0, c, false, false);
}

// ---------------------------------------------------------------------------
// Kernel 1: localization network.  blockDim = 32 (one wave), 16 images/block.
// ---------------------------------------------------------------------------
__global__ __launch_bounds__(32)
void stn_loc_kernel(const float* __restrict__ x,
                    const float* __restrict__ k1, const float* __restrict__ bc1,
                    const float* __restrict__ k2, const float* __restrict__ bc2,
                    const float* __restrict__ W1, const float* __restrict__ b1,
                    const float* __restrict__ W2, const float* __restrict__ b2,
                    float* __restrict__ theta_out)
{
  // LDS: ~41.5 KB per (single-wave) workgroup.
  __shared__ __align__(16) __bf16 wt1[16 * 64];    // conv1 W^T  [oc][tap]
  __shared__ __align__(16) __bf16 wt2[16 * 224];   // conv2 W^T  [oc][tap*8+ic]
  __shared__ __align__(16) __bf16 wt3[32 * 96];    // fc1   W^T  [n][k]
  __shared__ __align__(16) __bf16 wt4[16 * 32];    // fc2   W^T  [n][k]
  __shared__ __align__(16) __bf16 img[16 * 144];   // 16 images, 12x12
  __shared__ __align__(16) __bf16 col[16 * 224];   // im2col staging
  __shared__ __align__(16) __bf16 xs1[16 * 288];   // conv1 out 6x6x8
  __shared__ __align__(16) __bf16 xs2[16 * 96];    // conv2 out 3x3x10 pad->96
  __shared__ __align__(16) __bf16 h1 [16 * 32];    // fc1 out

  const int lane  = threadIdx.x;      // 0..31
  const int n     = lane & 15;        // N index / M index for A rows
  const int half  = lane >> 4;        // 0 or 1
  const int abase = half ? 16 : 0;    // A-fragment K sub-offset
  const int bbase = half ? 16 : 0;    // B-fragment K sub-offset
  const long img0 = (long)blockIdx.x * 16;
  const __bf16 bz = (__bf16)0.0f;

  // ---- weights -> LDS (bf16, transposed to [N][Kpad]); once per block ----
  for (int i = lane; i < 16 * 64; i += 32) {
    int oc = i >> 6, k = i & 63;
    float v = (oc < 8 && k < 49) ? k1[k * 8 + oc] : 0.f;   // (7,7,1,8)
    wt1[i] = f2bf(v);
  }
  for (int i = lane; i < 16 * 224; i += 32) {
    int oc = i / 224, k = i % 224;
    float v = (oc < 10 && k < 200) ? k2[k * 10 + oc] : 0.f; // (5,5,8,10)
    wt2[i] = f2bf(v);
  }
  for (int i = lane; i < 32 * 96; i += 32) {
    int nn = i / 96, k = i % 96;
    float v = (k < 90) ? W1[k * 32 + nn] : 0.f;             // (90,32)
    wt3[i] = f2bf(v);
  }
  for (int i = lane; i < 16 * 32; i += 32) {
    int nn = i >> 5, k = i & 31;
    float v = (nn < 6) ? W2[k * 6 + nn] : 0.f;              // (32,6)
    wt4[i] = f2bf(v);
  }
  // ---- image tile -> LDS bf16: float4 global loads, b64 LDS stores ----
  {
    const float4* xv = reinterpret_cast<const float4*>(x + img0 * 144);
    for (int i = lane; i < (16 * 144) / 4; i += 32) {
      float4 f = xv[i];
      v4bf p = { f2bf(f.x), f2bf(f.y), f2bf(f.z), f2bf(f.w) };
      *reinterpret_cast<v4bf*>(&img[i * 4]) = p;
    }
  }
  // ---- zero fc1-input pad columns (90..95) ----
  for (int i = lane; i < 16 * 96; i += 32)
    if ((i % 96) >= 90) xs2[i] = bz;
  lds_sync();

  const float bias1 = (n < 8)  ? bc1[n] : 0.f;
  const float bias2 = (n < 10) ? bc2[n] : 0.f;

  // Per-lane loop-invariant tap geometry: lane fills im2col columns
  // k = lane + 32*j for all 16 rows.  (dr,dc,ic,valid) depend only on k.
  int dr1[2], dc1[2]; bool ok1[2];
#pragma unroll
  for (int j = 0; j < 2; ++j) {
    int k = lane + 32 * j;
    ok1[j] = (k < 49);
    dr1[j] = k / 7 - 3;
    dc1[j] = k % 7 - 3;
  }
  int dr2[7], dc2[7], ic2[7]; bool ok2[7];
#pragma unroll
  for (int j = 0; j < 7; ++j) {
    int k = lane + 32 * j;
    ok2[j] = (k < 200);
    int t  = k >> 3;
    ic2[j] = k & 7;
    dr2[j] = t / 5 - 2;
    dc2[j] = t % 5 - 2;
  }

  // ================= conv1 (7x7, 1->8) + maxpool2 + relu =================
  for (int pr = 0; pr < 6; ++pr)
  for (int pc = 0; pc < 6; ++pc) {
    float best[8];
#pragma unroll
    for (int e = 0; e < 8; ++e) best[e] = -1e30f;
    for (int s = 0; s < 4; ++s) {
      const int r = pr * 2 + (s >> 1), c = pc * 2 + (s & 1);
      lds_sync();                                  // prior frag reads done
#pragma unroll
      for (int j = 0; j < 2; ++j) {                // build im2col [16][64]
        const int rr = r + dr1[j], cc = c + dc1[j];
        const bool inb = ok1[j] && (unsigned)rr < 12u && (unsigned)cc < 12u;
        const int base = inb ? (rr * 12 + cc) : 0;
        const int kcol = lane + 32 * j;
#pragma unroll
        for (int m = 0; m < 16; ++m) {
          __bf16 v = img[m * 144 + base];
          col[m * 64 + kcol] = inb ? v : bz;
        }
      }
      lds_sync();
      v8f acc = {};
#pragma unroll
      for (int kc = 0; kc < 64; kc += 32) {
        v16bf a = load_a(&col[n * 64], kc, abase);
        v16bf b = load_b(&wt1[n * 64], kc, bbase);
        acc = wmma_bf16(a, b, acc);
      }
#pragma unroll
      for (int e = 0; e < 8; ++e) best[e] = fmaxf(best[e], acc[e] + bias1);
    }
    if (n < 8) {
#pragma unroll
      for (int e = 0; e < 8; ++e) {
        int m = half * 8 + e;
        xs1[m * 288 + (pr * 6 + pc) * 8 + n] = f2bf(fmaxf(best[e], 0.f));
      }
    }
  }

  // ================= conv2 (5x5, 8->10) + maxpool2 + relu =================
  for (int qr = 0; qr < 3; ++qr)
  for (int qc = 0; qc < 3; ++qc) {
    float best[8];
#pragma unroll
    for (int e = 0; e < 8; ++e) best[e] = -1e30f;
    for (int s = 0; s < 4; ++s) {
      const int r = qr * 2 + (s >> 1), c = qc * 2 + (s & 1);
      lds_sync();
#pragma unroll
      for (int j = 0; j < 7; ++j) {                // build im2col [16][224]
        const int rr = r + dr2[j], cc = c + dc2[j];
        const bool inb = ok2[j] && (unsigned)rr < 6u && (unsigned)cc < 6u;
        const int base = inb ? ((rr * 6 + cc) * 8 + ic2[j]) : 0;
        const int kcol = lane + 32 * j;
#pragma unroll
        for (int m = 0; m < 16; ++m) {
          __bf16 v = xs1[m * 288 + base];
          col[m * 224 + kcol] = inb ? v : bz;
        }
      }
      lds_sync();
      v8f acc = {};
#pragma unroll
      for (int kc = 0; kc < 224; kc += 32) {
        v16bf a = load_a(&col[n * 224], kc, abase);
        v16bf b = load_b(&wt2[n * 224], kc, bbase);
        acc = wmma_bf16(a, b, acc);
      }
#pragma unroll
      for (int e = 0; e < 8; ++e) best[e] = fmaxf(best[e], acc[e] + bias2);
    }
    if (n < 10) {
#pragma unroll
      for (int e = 0; e < 8; ++e) {
        int m = half * 8 + e;
        xs2[m * 96 + (qr * 3 + qc) * 10 + n] = f2bf(fmaxf(best[e], 0.f));
      }
    }
  }
  lds_sync();

  // ================= fc1: (16x90)(90x32) + relu =================
  for (int t = 0; t < 2; ++t) {
    v8f acc = {};
#pragma unroll
    for (int kc = 0; kc < 96; kc += 32) {
      v16bf a = load_a(&xs2[n * 96], kc, abase);
      v16bf b = load_b(&wt3[(t * 16 + n) * 96], kc, bbase);
      acc = wmma_bf16(a, b, acc);
    }
    const float bb = b1[t * 16 + n];
#pragma unroll
    for (int e = 0; e < 8; ++e) {
      int m = half * 8 + e;
      h1[m * 32 + t * 16 + n] = f2bf(fmaxf(acc[e] + bb, 0.f));
    }
  }
  lds_sync();

  // ================= fc2: (16x32)(32x6) + b2, /(1+1e-5) =================
  {
    v8f acc = {};
    v16bf a = load_a(&h1[n * 32], 0, abase);
    v16bf b = load_b(&wt4[n * 32], 0, bbase);
    acc = wmma_bf16(a, b, acc);
    if (n < 6) {
      const float bb  = b2[n];
      const float inv = 1.0f / (1.0f + 1e-5f);
#pragma unroll
      for (int e = 0; e < 8; ++e) {
        int m = half * 8 + e;
        theta_out[(img0 + m) * 6 + n] = (acc[e] + bb) * inv;
      }
    }
  }
}

// ---------------------------------------------------------------------------
// Kernel 2: affine grid + bilinear sampling with scipy 'reflect' indexing.
// 288 threads = 2 images per block (144 pixels each).
// ---------------------------------------------------------------------------
__device__ __forceinline__ int reflect_idx(int i, int nsz) {
  // (d c b a | a b c d | d c b a)
  int m = 2 * nsz;
  i = ((i % m) + m) % m;
  return (i < nsz) ? i : (m - 1 - i);
}

__global__ __launch_bounds__(288)
void stn_sample_kernel(const float* __restrict__ x,
                       const float* __restrict__ theta,
                       float* __restrict__ out)
{
  __shared__ float th[12];
  const int tid = threadIdx.x;
  if (tid < 12) th[tid] = theta[(long)blockIdx.x * 12 + tid];
  __syncthreads();

  const int li = tid / 144;                 // image within block (0/1)
  const long b = (long)blockIdx.x * 2 + li;
  const int p  = tid - li * 144;
  const int i  = p / 12, j = p % 12;

  const float* t = &th[li * 6];
  const float xn = -1.f + (float)j * (2.f / 11.f);
  const float yn = -1.f + (float)i * (2.f / 11.f);
  const float sx = t[0] * xn + t[1] * yn + t[2];
  const float sy = t[3] * xn + t[4] * yn + t[5];
  const float px = (sx + 1.f) * 5.5f;
  const float py = (sy + 1.f) * 5.5f;

  const int x0 = (int)floorf(px), y0 = (int)floorf(py);
  const float fx = px - (float)x0, fy = py - (float)y0;
  const int x0r = reflect_idx(x0, 12),     x1r = reflect_idx(x0 + 1, 12);
  const int y0r = reflect_idx(y0, 12),     y1r = reflect_idx(y0 + 1, 12);

  const float* im = x + b * 144;
  const float v00 = im[y0r * 12 + x0r], v01 = im[y0r * 12 + x1r];
  const float v10 = im[y1r * 12 + x0r], v11 = im[y1r * 12 + x1r];
  out[b * 144 + p] =
      (1.f - fy) * ((1.f - fx) * v00 + fx * v01) +
      fy         * ((1.f - fx) * v10 + fx * v11);
}

// ---------------------------------------------------------------------------
extern "C" void kernel_launch(void* const* d_in, const int* in_sizes, int n_in,
                              void* d_out, int out_size, void* d_ws, size_t ws_size,
                              hipStream_t stream) {
  const float* x   = (const float*)d_in[0];
  const float* k1  = (const float*)d_in[1];
  const float* bc1 = (const float*)d_in[2];
  const float* k2  = (const float*)d_in[3];
  const float* bc2 = (const float*)d_in[4];
  const float* W1  = (const float*)d_in[5];
  const float* b1  = (const float*)d_in[6];
  const float* W2  = (const float*)d_in[7];
  const float* b2  = (const float*)d_in[8];

  const int B = in_sizes[0] / 144;          // 131072
  float* theta = (float*)d_ws;              // B*6 floats of scratch

  stn_loc_kernel<<<B / 16, 32, 0, stream>>>(x, k1, bc1, k2, bc2,
                                            W1, b1, W2, b2, theta);
  stn_sample_kernel<<<B / 2, 288, 0, stream>>>(x, theta, (float*)d_out);
}
